// Transformer_19138374271211
// MI455X (gfx1250) — matrix operand order repaired and checked by reading, not compiled, not run on hardware
//
#include <hip/hip_runtime.h>

// ---------------- model constants ----------------
#define MB   8       // batch
#define ML   128     // seq steps
#define MD   512     // model dim
#define MH   8       // heads
#define MDK  64      // head dim
#define MDFF 2048
#define MNL  6
#define MT   383
#define MTP  384     // padded seq
#define MSC  14
#define MAC  73
#define NS   (MB*ML)       // 1024 state images
#define NA   (MB*(ML-1))   // 1016 action images

// ---------------- vector types ----------------
typedef __attribute__((ext_vector_type(16))) __bf16 v16bf;
typedef __attribute__((ext_vector_type(8)))  __bf16 v8bf;
typedef __attribute__((ext_vector_type(4)))  __bf16 v4bf;
typedef __attribute__((ext_vector_type(8)))  float  v8f;

__device__ __forceinline__ __bf16 f2bf(float f) {
  union { float f; unsigned u; } x; x.f = f;
  unsigned r = x.u + 0x7FFFu + ((x.u >> 16) & 1u);  // round-to-nearest-even
  unsigned short h = (unsigned short)(r >> 16);
  union { unsigned short s; __bf16 b; } y; y.s = h; return y.b;
}

// ---------------- fp32 -> bf16 bulk convert (n multiple of 4) ----------------
__global__ void cvt4(const float* __restrict__ in, __bf16* __restrict__ outp, long n4) {
  long i = (long)blockIdx.x * blockDim.x + threadIdx.x;
  if (i >= n4) return;
  float4 f = ((const float4*)in)[i];
  v4bf o; o[0] = f2bf(f.x); o[1] = f2bf(f.y); o[2] = f2bf(f.z); o[3] = f2bf(f.w);
  ((v4bf*)outp)[i] = o;
}

// ---------------- WMMA GEMM (bf16 in, fp32 acc, fp32 and/or bf16 out) ----------------
// C = alpha * (A @ op(B)) [+ bias] [relu]
// BT=1: B is (N,K) row-major (x @ W^T).  BT=0: B is (K,N) row-major.
// One wave computes a 64x32 macro-tile (4 M-tiles x 2 N-tiles, 8 WMMAs/K-step).
// Loads are clamped in-bounds; garbage only reaches discarded output rows/cols.
// mode 0: plain (M,N) store at batch offset bz*sC
// mode 1: head-split store: row=b*MTP+t, col=h*64+d -> ((b*MH+h)*MTP+t)*MDK+d
// mode 2: head-merge store (batched bz=b*MH+h):      (b*MTP+row)*MD + h*MDK + col
template<int BT>
__global__ __launch_bounds__(32)
void gemm_wmma_bf16(const __bf16* __restrict__ A, const __bf16* __restrict__ B,
                    const float* __restrict__ bias,
                    float* __restrict__ Cf, __bf16* __restrict__ Ch,
                    int M, int N, int K,
                    long sA, long sB, long sC,
                    int mode, int relu, float alpha) {
  const int bz = blockIdx.z;
  const __bf16* Ab = A + (long)bz * sA;
  const __bf16* Bb = B + (long)bz * sB;
  const int lane = threadIdx.x;
  const int half = lane >> 4, l15 = lane & 15;
  const int m0 = blockIdx.y * 64;
  const int n0 = blockIdx.x * 32;

  int amr[4], bnr[2];
#pragma unroll
  for (int i = 0; i < 4; ++i) { int r = m0 + i * 16 + l15; amr[i] = r < M ? r : M - 1; }
#pragma unroll
  for (int j = 0; j < 2; ++j) { int c = n0 + j * 16 + l15; bnr[j] = c < N ? c : N - 1; }

  v8f zero = {};
  v8f acc[4][2];
#pragma unroll
  for (int i = 0; i < 4; ++i)
#pragma unroll
    for (int j = 0; j < 2; ++j) acc[i][j] = zero;

  for (int k0 = 0; k0 < K; k0 += 32) {
    // A fragment: lanes0-15 K = k0..k0+15 ; lanes16-31 K = k0+16..k0+31
    const int ab = k0 + half * 16;
    v16bf av[4];
#pragma unroll
    for (int i = 0; i < 4; ++i) {
      const __bf16* Ar = Ab + (long)amr[i] * K + ab;
      v8bf lo = *(const v8bf*)(Ar);
      v8bf hi = *(const v8bf*)(Ar + 8);
      av[i] = __builtin_shufflevector(lo, hi, 0,1,2,3,4,5,6,7,8,9,10,11,12,13,14,15);
    }
    // B fragment: lanes0-15 K = k0..k0+15 ; lanes16-31 K = k0+16..k0+31
    const int kb = k0 + half * 16;
    v16bf bv[2];
#pragma unroll
    for (int j = 0; j < 2; ++j) {
      if (BT) {
        const __bf16* Br = Bb + (long)bnr[j] * K + kb;   // (N,K): 16 contiguous
        v8bf lo = *(const v8bf*)(Br);
        v8bf hi = *(const v8bf*)(Br + 8);
        bv[j] = __builtin_shufflevector(lo, hi, 0,1,2,3,4,5,6,7,8,9,10,11,12,13,14,15);
      } else {
        v16bf t;                                         // (K,N): strided column
#pragma unroll
        for (int e = 0; e < 16; ++e)
          t[e] = Bb[(long)(kb + e) * N + bnr[j]];
        bv[j] = t;
      }
    }
#pragma unroll
    for (int i = 0; i < 4; ++i)
#pragma unroll
      for (int j = 0; j < 2; ++j)
        acc[i][j] = __builtin_amdgcn_wmma_f32_16x16x32_bf16(
            false, av[i], false, bv[j], (short)0, acc[i][j], false, false);
  }

#pragma unroll
  for (int i = 0; i < 4; ++i) {
#pragma unroll
    for (int j = 0; j < 2; ++j) {
#pragma unroll
      for (int r = 0; r < 8; ++r) {
        const int row = m0 + i * 16 + half * 8 + r;   // D: VGPR r -> M=r (+8 hi lanes)
        const int col = n0 + j * 16 + l15;
        if (row < M && col < N) {
          float v = acc[i][j][r] * alpha;
          if (bias) v += bias[col];
          if (relu) v = fmaxf(v, 0.f);
          long gidx;
          if (mode == 0) {
            gidx = bz * sC + (long)row * N + col;
          } else if (mode == 1) {
            int bb = row / MTP, tt = row - bb * MTP, hh = col >> 6, dd = col & 63;
            gidx = (((long)bb * MH + hh) * MTP + tt) * MDK + dd;
          } else {
            int bb = bz >> 3, hh = bz & 7;
            gidx = ((long)bb * MTP + row) * MD + hh * MDK + col;
          }
          if (Cf) Cf[gidx] = v;
          if (Ch) Ch[gidx] = f2bf(v);
        }
      }
    }
  }
}

// ---------------- direct 3x3 conv, pad 1, 8x8, + ReLU (bf16 io, fp32 acc) ----------------
__global__ void conv3x3_relu(const __bf16* __restrict__ x, const __bf16* __restrict__ w,
                             const float* __restrict__ b, __bf16* __restrict__ y,
                             int nimg, int IC, int OC) {
  int idx = blockIdx.x * blockDim.x + threadIdx.x;
  int total = nimg * OC * 64;
  if (idx >= total) return;
  int px = idx & 63, oc = (idx >> 6) % OC, img = idx / (OC * 64);
  int oy = px >> 3, ox = px & 7;
  const __bf16* xi = x + (long)img * IC * 64;
  const __bf16* wo = w + (long)oc * IC * 9;
  float acc = b[oc];
  for (int ic = 0; ic < IC; ++ic) {
    const __bf16* xc = xi + ic * 64;
    const __bf16* wc = wo + ic * 9;
#pragma unroll
    for (int ky = 0; ky < 3; ++ky) {
      int iy = oy + ky - 1; if ((unsigned)iy >= 8u) continue;
#pragma unroll
      for (int kx = 0; kx < 3; ++kx) {
        int ix = ox + kx - 1; if ((unsigned)ix >= 8u) continue;
        acc += (float)xc[iy * 8 + ix] * (float)wc[ky * 3 + kx];
      }
    }
  }
  y[idx] = f2bf(fmaxf(acc, 0.f));
}

// ---------------- transposed conv 3x3 (lhs_dilation s, pad (1,1+op)) ----------------
// w is the ORIGINAL (IC, OC, 3, 3) in bf16; flipped+transposed taps like reference _convT
__global__ void convT3x3(const __bf16* __restrict__ x, const __bf16* __restrict__ w,
                         const float* __restrict__ b,
                         float* __restrict__ yf, __bf16* __restrict__ yh,
                         int nimg, int IC, int OC, int inH, int outH, int s, int relu) {
  int idx = blockIdx.x * blockDim.x + threadIdx.x;
  int total = nimg * OC * outH * outH;
  if (idx >= total) return;
  int ox = idx % outH; int t = idx / outH;
  int oy = t % outH;   t /= outH;
  int oc = t % OC;     int img = t / OC;
  const __bf16* xi = x + (long)img * IC * inH * inH;
  float acc = b[oc];
  int dil = (inH - 1) * s + 1;
  for (int ic = 0; ic < IC; ++ic) {
    const __bf16* wv = w + ((long)ic * OC + oc) * 9;
    const __bf16* xc = xi + ic * inH * inH;
#pragma unroll
    for (int ky = 0; ky < 3; ++ky) {
      int m = oy + ky - 1; if (m < 0 || m >= dil || (m % s)) continue;
#pragma unroll
      for (int kx = 0; kx < 3; ++kx) {
        int n = ox + kx - 1; if (n < 0 || n >= dil || (n % s)) continue;
        acc += (float)xc[(m / s) * inH + (n / s)] * (float)wv[(2 - ky) * 3 + (2 - kx)];
      }
    }
  }
  if (relu) acc = fmaxf(acc, 0.f);
  if (yf) yf[idx] = acc;
  else    yh[idx] = f2bf(acc);
}

// ---------------- first MLP-encoder layer (K=1): h = relu(x*w1 + b1) -> bf16 ----------------
__global__ void enc_k1(const float* __restrict__ xf, const int* __restrict__ xi,
                       const float* __restrict__ w1, const float* __restrict__ b1,
                       __bf16* __restrict__ h, int M, int N) {
  int idx = blockIdx.x * blockDim.x + threadIdx.x;
  if (idx >= M * N) return;
  int n = idx % N, m = idx / N;
  float xv = xf ? xf[m] : (float)xi[m];
  h[idx] = f2bf(fmaxf(xv * w1[n] + b1[n], 0.f));
}

// ---------------- layernorm over D=512 (optional residual), fp32 + optional bf16 out ----------------
__global__ void ln_kernel(const float* __restrict__ xin, const float* __restrict__ res,
                          const float* __restrict__ w, const float* __restrict__ b,
                          float* __restrict__ y, __bf16* __restrict__ ybf) {
  int row = blockIdx.x, tid = threadIdx.x;      // 256 threads
  const float* xr = xin + (long)row * MD;
  const float* rr = res ? res + (long)row * MD : nullptr;
  __shared__ float buf[MD];
  __shared__ float red[256];
  float s = 0.f;
  for (int d = tid; d < MD; d += 256) { float v = xr[d] + (rr ? rr[d] : 0.f); buf[d] = v; s += v; }
  red[tid] = s; __syncthreads();
  for (int st = 128; st > 0; st >>= 1) { if (tid < st) red[tid] += red[tid + st]; __syncthreads(); }
  float m = red[0] / (float)MD; __syncthreads();
  float s2 = 0.f;
  for (int d = tid; d < MD; d += 256) { float t = buf[d] - m; s2 += t * t; }
  red[tid] = s2; __syncthreads();
  for (int st = 128; st > 0; st >>= 1) { if (tid < st) red[tid] += red[tid + st]; __syncthreads(); }
  float inv = rsqrtf(red[0] / (float)MD + 1e-5f);
  for (int d = tid; d < MD; d += 256) {
    float v = (buf[d] - m) * inv * w[d] + b[d];
    y[(long)row * MD + d] = v;
    if (ybf) ybf[(long)row * MD + d] = f2bf(v);
  }
}

// ---------------- causal masked softmax on bf16 rows of 384 (fp32 math) ----------------
__global__ void softmax_mask(__bf16* __restrict__ sc) {
  int rid = blockIdx.x, tid = threadIdx.x;      // 128 threads; rid = (b*H+h)*MTP + q
  int qi = rid % MTP;
  int lim = qi < MT ? qi : MT - 1;              // keys k <= q, k < 383
  __bf16* row = sc + (long)rid * MTP;
  __shared__ float red[128];
  __shared__ float e_s[MTP];
  float mx = -1e30f;
  for (int k = tid; k < MTP; k += 128) if (k <= lim) mx = fmaxf(mx, (float)row[k]);
  red[tid] = mx; __syncthreads();
  for (int st = 64; st > 0; st >>= 1) { if (tid < st) red[tid] = fmaxf(red[tid], red[tid + st]); __syncthreads(); }
  mx = red[0]; __syncthreads();
  float s = 0.f;
  for (int k = tid; k < MTP; k += 128) {
    float e = (k <= lim) ? __expf((float)row[k] - mx) : 0.f;
    e_s[k] = e; s += e;
  }
  red[tid] = s; __syncthreads();
  for (int st = 64; st > 0; st >>= 1) { if (tid < st) red[tid] += red[tid + st]; __syncthreads(); }
  float inv = 1.f / red[0];
  for (int k = tid; k < MTP; k += 128) row[k] = f2bf(e_s[k] * inv);
}

// ---------------- interleave embeddings into padded sequence (-> bf16) ----------------
__global__ void comb_assemble(const float* __restrict__ r_emb, const float* __restrict__ s_emb,
                              const float* __restrict__ a_emb, const float* __restrict__ t_emb,
                              __bf16* __restrict__ comb) {
  long idx = (long)blockIdx.x * blockDim.x + threadIdx.x;
  long total = (long)MB * MTP * MD;
  if (idx >= total) return;
  int d = idx & (MD - 1);
  int t = (int)((idx >> 9) % MTP);
  int b = (int)(idx / ((long)MTP * MD));
  float v = 0.f;
  if (t < MT) {
    int i = t / 3, r = t % 3;
    float te = t_emb[((long)b * ML + i) * MD + d];
    if (r == 0)      v = r_emb[((long)b * ML + i) * MD + d] + te;
    else if (r == 1) v = s_emb[((long)b * ML + i) * MD + d] + te;
    else             v = a_emb[((long)b * (ML - 1) + i) * MD + d] + te;
  }
  comb[idx] = f2bf(v);
}

// ---------------- gather every-3rd token into dense bf16 (1024, 512) ----------------
__global__ void gather_tokens(const float* __restrict__ x, __bf16* __restrict__ z, int offset) {
  long idx = (long)blockIdx.x * blockDim.x + threadIdx.x;
  if (idx >= (long)NS * MD) return;
  int d = idx & (MD - 1);
  int r = (int)(idx >> 9);
  int b = r >> 7, i = r & 127;
  z[idx] = f2bf(x[((long)b * MTP + offset + 3 * i) * MD + d]);
}

// ---------------- reward head: dot(row, w) + b ----------------
__global__ void reward_head(const float* __restrict__ x, const float* __restrict__ w,
                            const float* __restrict__ bias, float* __restrict__ out) {
  int row = blockIdx.x, tid = threadIdx.x;      // 128 threads, rows = 8*127
  int b = row / (ML - 1), i = row % (ML - 1);
  const float* xr = x + ((long)b * MTP + 2 + 3 * i) * MD;
  __shared__ float red[128];
  float s = 0.f;
  for (int d = tid; d < MD; d += 128) s += xr[d] * w[d];
  red[tid] = s; __syncthreads();
  for (int st = 64; st > 0; st >>= 1) { if (tid < st) red[tid] += red[tid + st]; __syncthreads(); }
  if (tid == 0) out[row] = red[0] + bias[0];
}

// =====================================================================
extern "C" void kernel_launch(void* const* d_in, const int* in_sizes, int n_in,
                              void* d_out, int out_size, void* d_ws, size_t ws_size,
                              hipStream_t stream) {
  (void)in_sizes; (void)n_in; (void)out_size; (void)ws_size;
  const float* states  = (const float*)d_in[0];
  const float* rtgs    = (const float*)d_in[1];
  const int*   tsteps  = (const int*)d_in[2];
  const float* actions = (const float*)d_in[3];
  // d_in[4] = mask (implemented analytically)
  const float *se_c1w=(const float*)d_in[5],  *se_c1b=(const float*)d_in[6],
              *se_c2w=(const float*)d_in[7],  *se_c2b=(const float*)d_in[8],
              *se_c3w=(const float*)d_in[9],  *se_c3b=(const float*)d_in[10],
              *se_fcw=(const float*)d_in[11], *se_fcb=(const float*)d_in[12],
              *se_lnw=(const float*)d_in[13], *se_lnb=(const float*)d_in[14];
  const float *ae_c1w=(const float*)d_in[15], *ae_c1b=(const float*)d_in[16],
              *ae_c2w=(const float*)d_in[17], *ae_c2b=(const float*)d_in[18],
              *ae_c3w=(const float*)d_in[19], *ae_c3b=(const float*)d_in[20],
              *ae_fcw=(const float*)d_in[21], *ae_fcb=(const float*)d_in[22],
              *ae_lnw=(const float*)d_in[23], *ae_lnb=(const float*)d_in[24];
  const float *rtg_w1=(const float*)d_in[25], *rtg_b1=(const float*)d_in[26],
              *rtg_w2=(const float*)d_in[27], *rtg_b2=(const float*)d_in[28],
              *rtg_lnw=(const float*)d_in[29],*rtg_lnb=(const float*)d_in[30];
  const float *ts_w1=(const float*)d_in[31],  *ts_b1=(const float*)d_in[32],
              *ts_w2=(const float*)d_in[33],  *ts_b2=(const float*)d_in[34],
              *ts_lnw=(const float*)d_in[35], *ts_lnb=(const float*)d_in[36];
  const float *ip_w=(const float*)d_in[37],   *ip_b=(const float*)d_in[38];
  const float *lyr_wq=(const float*)d_in[39], *lyr_bq=(const float*)d_in[40],
              *lyr_wk=(const float*)d_in[41], *lyr_bk=(const float*)d_in[42],
              *lyr_wv=(const float*)d_in[43], *lyr_bv=(const float*)d_in[44],
              *lyr_wo=(const float*)d_in[45], *lyr_bo=(const float*)d_in[46],
              *lyr_f1w=(const float*)d_in[47],*lyr_f1b=(const float*)d_in[48],
              *lyr_f2w=(const float*)d_in[49],*lyr_f2b=(const float*)d_in[50],
              *lyr_l1w=(const float*)d_in[51],*lyr_l1b=(const float*)d_in[52],
              *lyr_l2w=(const float*)d_in[53],*lyr_l2b=(const float*)d_in[54];
  const float *ad_fcw=(const float*)d_in[55], *ad_fcb=(const float*)d_in[56],
              *ad_w1=(const float*)d_in[57],  *ad_b1=(const float*)d_in[58],
              *ad_w2=(const float*)d_in[59],  *ad_b2=(const float*)d_in[60],
              *ad_w3=(const float*)d_in[61],  *ad_b3=(const float*)d_in[62];
  const float *sd_fcw=(const float*)d_in[63], *sd_fcb=(const float*)d_in[64],
              *sd_w1=(const float*)d_in[65],  *sd_b1=(const float*)d_in[66],
              *sd_w2=(const float*)d_in[67],  *sd_b2=(const float*)d_in[68],
              *sd_w3=(const float*)d_in[69],  *sd_b3=(const float*)d_in[70];
  const float *rp_w=(const float*)d_in[71],   *rp_b=(const float*)d_in[72];

  float* out = (float*)d_out;
  const long OUT_ACT = (long)MB * ML * MAC * 64;   // 4,784,128
  const long OUT_ST  = (long)MB * ML * MSC * 64;   //   917,504

  // ---------- workspace layout ----------
  char* base = (char*)d_ws;
  size_t off = 0;
  auto allocB = [&](size_t bytes) { off = (off + 255) & ~(size_t)255; void* p = base + off; off += bytes; return p; };
  auto allocF = [&](size_t n) { return (float*)allocB(n * 4); };
  auto allocH = [&](size_t n) { return (__bf16*)allocB(n * 2); };

  // bf16 weights (converted once per launch)
  __bf16* se_fcw_b = allocH(2097152);
  __bf16* ae_fcw_b = allocH(2097152);
  __bf16* rtg_w2_b = allocH(262144);
  __bf16* ts_w2_b  = allocH(262144);
  __bf16* ip_w_b   = allocH(262144);
  __bf16* wq_b = allocH(6*262144);
  __bf16* wk_b = allocH(6*262144);
  __bf16* wv_b = allocH(6*262144);
  __bf16* wo_b = allocH(6*262144);
  __bf16* f1w_b = allocH(6*1048576);
  __bf16* f2w_b = allocH(6*1048576);
  __bf16* ad_fcw_b = allocH(262144);
  __bf16* sd_fcw_b = allocH(262144);
  __bf16* se_c1w_b = allocH(4032);   __bf16* se_c2w_b = allocH(18432);  __bf16* se_c3w_b = allocH(36864);
  __bf16* ae_c1w_b = allocH(21024);  __bf16* ae_c2w_b = allocH(18432);  __bf16* ae_c3w_b = allocH(36864);
  __bf16* ad_w1_b  = allocH(73728);  __bf16* ad_w2_b  = allocH(18432);  __bf16* ad_w3_b  = allocH(21024);
  __bf16* sd_w1_b  = allocH(73728);  __bf16* sd_w2_b  = allocH(18432);  __bf16* sd_w3_b  = allocH(4032);
  // bf16 activations
  __bf16* scoresb = allocH((size_t)MB*MH*MTP*MTP);   // 9,437,184 (also conv ping/pong + dec stages)
  __bf16* qpb = allocH((size_t)MB*MH*MTP*MDK);
  __bf16* kpb = allocH((size_t)MB*MH*MTP*MDK);
  __bf16* vpb = allocH((size_t)MB*MH*MTP*MDK);
  __bf16* xbf = allocH((size_t)MB*MTP*MD);
  __bf16* cbf = allocH((size_t)MB*MTP*MD);           // comb / ctx / enc_k1 scratch
  __bf16* ffhb = allocH((size_t)MB*MTP*MDFF);        // FFN hidden; also raw-input bf16 at start
  __bf16* zab = allocH((size_t)NS*MD);
  __bf16* zsb = allocH((size_t)NS*MD);
  __bf16* dfcb = allocH((size_t)NS*MD);
  // fp32 buffers
  float* x     = allocF((size_t)MB*MTP*MD);
  float* proj  = allocF((size_t)MB*MTP*MD);
  float* s_emb = allocF((size_t)NS*MD);
  float* a_emb = allocF((size_t)NS*MD);
  float* r_emb = allocF((size_t)NS*MD);
  float* t_emb = allocF((size_t)NS*MD);
  // overlays
  __bf16* convAb = scoresb;                 // 4,194,304 elems
  __bf16* convBb = scoresb + 4194304;       // 4,194,304 elems
  __bf16* statesb  = ffhb;                  //   917,504 elems
  __bf16* actionsb = ffhb + 917504;         // 4,746,752 elems

  auto cdiv = [](long n, long d) { return (unsigned)((n + d - 1) / d); };
  auto cvt = [&](const float* src, __bf16* dst, long n) {
    cvt4<<<cdiv(n / 4, 256), 256, 0, stream>>>(src, dst, n / 4);
  };
  auto gemm = [&](int bT, int mode, int relu, float alpha,
                  const __bf16* A, const __bf16* Bm, const float* bias,
                  float* Cf, __bf16* Ch,
                  int M, int N, int K, int batch, long sA, long sB, long sC) {
    dim3 g((N + 31) / 32, (M + 63) / 64, batch), blk(32, 1, 1);
    if (bT) gemm_wmma_bf16<1><<<g, blk, 0, stream>>>(A, Bm, bias, Cf, Ch, M, N, K, sA, sB, sC, mode, relu, alpha);
    else    gemm_wmma_bf16<0><<<g, blk, 0, stream>>>(A, Bm, bias, Cf, Ch, M, N, K, sA, sB, sC, mode, relu, alpha);
  };

  // ---------- one-time conversions ----------
  cvt(se_fcw, se_fcw_b, 2097152);  cvt(ae_fcw, ae_fcw_b, 2097152);
  cvt(rtg_w2, rtg_w2_b, 262144);   cvt(ts_w2, ts_w2_b, 262144);   cvt(ip_w, ip_w_b, 262144);
  cvt(lyr_wq, wq_b, 6*262144);     cvt(lyr_wk, wk_b, 6*262144);
  cvt(lyr_wv, wv_b, 6*262144);     cvt(lyr_wo, wo_b, 6*262144);
  cvt(lyr_f1w, f1w_b, 6*1048576);  cvt(lyr_f2w, f2w_b, 6*1048576);
  cvt(ad_fcw, ad_fcw_b, 262144);   cvt(sd_fcw, sd_fcw_b, 262144);
  cvt(se_c1w, se_c1w_b, 4032);   cvt(se_c2w, se_c2w_b, 18432);  cvt(se_c3w, se_c3w_b, 36864);
  cvt(ae_c1w, ae_c1w_b, 21024);  cvt(ae_c2w, ae_c2w_b, 18432);  cvt(ae_c3w, ae_c3w_b, 36864);
  cvt(ad_w1, ad_w1_b, 73728);    cvt(ad_w2, ad_w2_b, 18432);    cvt(ad_w3, ad_w3_b, 21024);
  cvt(sd_w1, sd_w1_b, 73728);    cvt(sd_w2, sd_w2_b, 18432);    cvt(sd_w3, sd_w3_b, 4032);
  cvt(states, statesb, (long)NS*MSC*64);
  cvt(actions, actionsb, (long)NA*MAC*64);

  // ---------- encoders ----------
  conv3x3_relu<<<cdiv((long)NS*32*64,256),256,0,stream>>>(statesb, se_c1w_b, se_c1b, convAb, NS, MSC, 32);
  conv3x3_relu<<<cdiv((long)NS*64*64,256),256,0,stream>>>(convAb,  se_c2w_b, se_c2b, convBb, NS, 32, 64);
  conv3x3_relu<<<cdiv((long)NS*64*64,256),256,0,stream>>>(convBb,  se_c3w_b, se_c3b, convAb, NS, 64, 64);
  gemm(1,0,0,1.f, convAb, se_fcw_b, se_fcb, s_emb, nullptr, NS, MD, 4096, 1,0,0,0);
  ln_kernel<<<NS,256,0,stream>>>(s_emb, nullptr, se_lnw, se_lnb, s_emb, nullptr);
  conv3x3_relu<<<cdiv((long)NA*32*64,256),256,0,stream>>>(actionsb, ae_c1w_b, ae_c1b, convAb, NA, MAC, 32);
  conv3x3_relu<<<cdiv((long)NA*64*64,256),256,0,stream>>>(convAb,   ae_c2w_b, ae_c2b, convBb, NA, 32, 64);
  conv3x3_relu<<<cdiv((long)NA*64*64,256),256,0,stream>>>(convBb,   ae_c3w_b, ae_c3b, convAb, NA, 64, 64);
  gemm(1,0,0,1.f, convAb, ae_fcw_b, ae_fcb, a_emb, nullptr, NA, MD, 4096, 1,0,0,0);
  ln_kernel<<<NA,256,0,stream>>>(a_emb, nullptr, ae_lnw, ae_lnb, a_emb, nullptr);
  // rtg / timestep MLPs
  enc_k1<<<cdiv((long)NS*MD,256),256,0,stream>>>(rtgs, nullptr, rtg_w1, rtg_b1, cbf, NS, MD);
  gemm(1,0,0,1.f, cbf, rtg_w2_b, rtg_b2, r_emb, nullptr, NS, MD, MD, 1,0,0,0);
  ln_kernel<<<NS,256,0,stream>>>(r_emb, nullptr, rtg_lnw, rtg_lnb, r_emb, nullptr);
  enc_k1<<<cdiv((long)NS*MD,256),256,0,stream>>>(nullptr, tsteps, ts_w1, ts_b1, cbf, NS, MD);
  gemm(1,0,0,1.f, cbf, ts_w2_b, ts_b2, t_emb, nullptr, NS, MD, MD, 1,0,0,0);
  ln_kernel<<<NS,256,0,stream>>>(t_emb, nullptr, ts_lnw, ts_lnb, t_emb, nullptr);

  // ---------- interleave + input projection (dual fp32 + bf16 x) ----------
  comb_assemble<<<cdiv((long)MB*MTP*MD,256),256,0,stream>>>(r_emb, s_emb, a_emb, t_emb, cbf);
  gemm(1,0,0,1.f, cbf, ip_w_b, ip_b, x, xbf, MB*MTP, MD, MD, 1,0,0,0);

  // ---------- transformer layers ----------
  const int Mrows = MB * MTP;          // 3072
  for (int l = 0; l < MNL; ++l) {
    const __bf16 *wq = wq_b + (size_t)l*MD*MD, *wk = wk_b + (size_t)l*MD*MD,
                 *wv = wv_b + (size_t)l*MD*MD, *wo = wo_b + (size_t)l*MD*MD,
                 *f1w = f1w_b + (size_t)l*MDFF*MD, *f2w = f2w_b + (size_t)l*MD*MDFF;
    const float *bq = lyr_bq + (size_t)l*MD, *bk = lyr_bk + (size_t)l*MD,
                *bv = lyr_bv + (size_t)l*MD, *bo = lyr_bo + (size_t)l*MD,
                *f1b = lyr_f1b + (size_t)l*MDFF, *f2b = lyr_f2b + (size_t)l*MD,
                *l1w = lyr_l1w + (size_t)l*MD, *l1b = lyr_l1b + (size_t)l*MD,
                *l2w = lyr_l2w + (size_t)l*MD, *l2b = lyr_l2b + (size_t)l*MD;

    // QKV (bf16 head-split)
    gemm(1,1,0,1.f, xbf, wq, bq, nullptr, qpb, Mrows, MD, MD, 1,0,0,0);
    gemm(1,1,0,1.f, xbf, wk, bk, nullptr, kpb, Mrows, MD, MD, 1,0,0,0);
    gemm(1,1,0,1.f, xbf, wv, bv, nullptr, vpb, Mrows, MD, MD, 1,0,0,0);
    // scores = (Q @ K^T)/8, batched over B*H, bf16 out
    gemm(1,0,0,0.125f, qpb, kpb, nullptr, nullptr, scoresb, MTP, MTP, MDK,
         MB*MH, (long)MTP*MDK, (long)MTP*MDK, (long)MTP*MTP);
    softmax_mask<<<MB*MH*MTP,128,0,stream>>>(scoresb);
    // ctx = att @ V (NN), head-merge into bf16 (3072,512)
    gemm(0,2,0,1.f, scoresb, vpb, nullptr, nullptr, cbf, MTP, MDK, MTP,
         MB*MH, (long)MTP*MTP, (long)MTP*MDK, 0);
    // o-projection -> fp32 proj; x = LN(x + proj) (dual out)
    gemm(1,0,0,1.f, cbf, wo, bo, proj, nullptr, Mrows, MD, MD, 1,0,0,0);
    ln_kernel<<<Mrows,256,0,stream>>>(x, proj, l1w, l1b, x, xbf);
    // FFN
    gemm(1,0,1,1.f, xbf, f1w, f1b, nullptr, ffhb, Mrows, MDFF, MD, 1,0,0,0);
    gemm(1,0,0,1.f, ffhb, f2w, f2b, proj, nullptr, Mrows, MD, MDFF, 1,0,0,0);
    ln_kernel<<<Mrows,256,0,stream>>>(x, proj, l2w, l2b, x, xbf);
  }

  // ---------- heads ----------
  gather_tokens<<<cdiv((long)NS*MD,256),256,0,stream>>>(x, zsb, 0);  // x_state  = x[:, ::3]
  gather_tokens<<<cdiv((long)NS*MD,256),256,0,stream>>>(x, zab, 1);  // x_action = x[:,1::3]

  __bf16* d1b = scoresb;                 // (1024,64,4,4) = 1,048,576
  __bf16* d2b = scoresb + 2097152;       // (1024,32,8,8) = 2,097,152
  // action decoder -> out[0 : OUT_ACT]
  gemm(1,0,1,1.f, zab, ad_fcw_b, ad_fcb, nullptr, dfcb, NS, MD, MD, 1,0,0,0);
  convT3x3<<<cdiv((long)NS*64*16,256),256,0,stream>>>(dfcb, ad_w1_b, ad_b1, nullptr, d1b, NS, 128, 64, 2, 4, 2, 1);
  convT3x3<<<cdiv((long)NS*32*64,256),256,0,stream>>>(d1b,  ad_w2_b, ad_b2, nullptr, d2b, NS, 64, 32, 4, 8, 2, 1);
  convT3x3<<<cdiv((long)NS*MAC*64,256),256,0,stream>>>(d2b, ad_w3_b, ad_b3, out, nullptr, NS, 32, MAC, 8, 8, 1, 0);
  // state decoder -> out[OUT_ACT : OUT_ACT+OUT_ST]
  gemm(1,0,1,1.f, zsb, sd_fcw_b, sd_fcb, nullptr, dfcb, NS, MD, MD, 1,0,0,0);
  convT3x3<<<cdiv((long)NS*64*16,256),256,0,stream>>>(dfcb, sd_w1_b, sd_b1, nullptr, d1b, NS, 128, 64, 2, 4, 2, 1);
  convT3x3<<<cdiv((long)NS*32*64,256),256,0,stream>>>(d1b,  sd_w2_b, sd_b2, nullptr, d2b, NS, 64, 32, 4, 8, 2, 1);
  convT3x3<<<cdiv((long)NS*MSC*64,256),256,0,stream>>>(d2b, sd_w3_b, sd_b3, out + OUT_ACT, nullptr, NS, 32, MSC, 8, 8, 1, 0);
  // reward head -> out[OUT_ACT+OUT_ST : +1016]
  reward_head<<<MB*(ML-1),128,0,stream>>>(x, rp_w, rp_b, out + OUT_ACT + OUT_ST);
}
